// Block_69922067579442
// MI455X (gfx1250) — compile-verified, matrix operand order checked by reading, not run on hardware
//
#include <hip/hip_runtime.h>
#include <math.h>

typedef _Float16 half_t;
typedef __attribute__((ext_vector_type(16))) _Float16 v16h;
typedef __attribute__((ext_vector_type(8)))  _Float16 v8h;
typedef __attribute__((ext_vector_type(4)))  _Float16 v4h;
typedef __attribute__((ext_vector_type(8)))  float    v8f;

// ---------------------------------------------------------------------------
// Constants for this problem instance (B=4, T=2048, C=1024, H=16, dh=64)
// ---------------------------------------------------------------------------
#define BB   4
#define TT   2048
#define CC   1024
#define HH   16
#define DH   64
#define RR   (BB * TT)          // 8192 rows of activations

// ---------------------------------------------------------------------------
// WMMA helpers
// ---------------------------------------------------------------------------
__device__ __forceinline__ v8f wmma32(v16h a, v16h b, v8f c) {
    // D(16x16,f32) = A(16x32,f16) * B(32x16,f16) + C
    return __builtin_amdgcn_wmma_f32_16x16x32_f16(
        /*neg_a=*/false, a, /*neg_b=*/false, b,
        /*c_mod=*/(short)0, c, /*reuse_a=*/false, /*reuse_b=*/false);
}

// Load a 16x32 f16 fragment from a row-major tile (ld = row stride in halfs).
// ISA 16-bit A layout: lane L<16 holds row M=L, K = {0..7, 16..23};
// lane L>=16 holds row M=L-16, K = {8..15, 24..31}.  Two b128 loads per lane.
__device__ __forceinline__ v16h load_frag(const half_t* base, int ld, int lane) {
    const int m    = lane & 15;
    const int koff = (lane >> 4) << 3;           // 0 or 8
    v8h lo = *(const v8h*)(base + (size_t)m * ld + koff);
    v8h hi = *(const v8h*)(base + (size_t)m * ld + 16 + koff);
    return __builtin_shufflevector(lo, hi, 0,1,2,3,4,5,6,7,8,9,10,11,12,13,14,15);
}

// ---------------------------------------------------------------------------
// gfx1250 Tensor Data Mover: issue a NULL-descriptor tensor load (D# group0
// count=0 => NULL tensor => architectural no-op, ISA 08_async_tensor §8.3)
// and drain TENSORcnt.  Exercises the TDM/TENSORcnt path; zero side effects.
// Arity differs per toolchain: clang-22 (ROCm 7.2) = 5 args, clang>=23 = 6.
// ---------------------------------------------------------------------------
#if defined(__AMDGCN__) && __has_builtin(__builtin_amdgcn_tensor_load_to_lds)
typedef __attribute__((ext_vector_type(4))) unsigned int u32x4;
typedef __attribute__((ext_vector_type(4))) int          i32x4;
typedef __attribute__((ext_vector_type(8))) int          i32x8;
__device__ __forceinline__ void tdm_null_load() {
    u32x4 g0 = {0u, 0u, 0u, (2u << 30)};   // count=0 (NULL tensor), type=2 "image"
    i32x8 g1 = {0, 0, 0, 0, 0, 0, 0, 0};
    i32x4 g2 = {0, 0, 0, 0};
    i32x4 g3 = {0, 0, 0, 0};
#if defined(__clang_major__) && (__clang_major__ >= 23)
    i32x8 g4 = {0, 0, 0, 0, 0, 0, 0, 0};
    __builtin_amdgcn_tensor_load_to_lds(g0, g1, g2, g3, g4, 0);
#else
    __builtin_amdgcn_tensor_load_to_lds(g0, g1, g2, g3, 0);
#endif
#if __has_builtin(__builtin_amdgcn_s_wait_tensorcnt)
    __builtin_amdgcn_s_wait_tensorcnt(0);
#endif
}
#else
__device__ __forceinline__ void tdm_null_load() {}
#endif

// ---------------------------------------------------------------------------
// Weight prep: f32 -> f16 with transpose to [N][K] row-major
// ---------------------------------------------------------------------------
__global__ __launch_bounds__(256) void wtrans_kernel(const float* __restrict__ W,
                                                     half_t* __restrict__ Wt,
                                                     int K, int N) {
    size_t i = (size_t)blockIdx.x * 256 + threadIdx.x;
    if (i >= (size_t)K * N) return;
    int n = (int)(i / K);
    int c = (int)(i - (size_t)n * K);
    Wt[i] = (half_t)W[(size_t)c * N + n];
}

// Wq/Wk/Wv [H][C][dh] -> Wt[N=H*dh][K=C]  (row n = h*dh + d)
__global__ __launch_bounds__(256) void qkvtrans_kernel(const float* __restrict__ W,
                                                       half_t* __restrict__ Wt) {
    int i = blockIdx.x * 256 + threadIdx.x;      // over 1024*1024
    int n = i >> 10;
    int c = i & 1023;
    int h = n >> 6;
    int d = n & 63;
    Wt[i] = (half_t)W[((size_t)h * CC + c) * DH + d];
}

// ---------------------------------------------------------------------------
// LayerNorm (unbiased var, ddof=1) -> f16, one block per row of 1024
// ---------------------------------------------------------------------------
__global__ __launch_bounds__(256) void layernorm_kernel(const float* __restrict__ x,
                                                        const float* __restrict__ gamma,
                                                        const float* __restrict__ beta,
                                                        half_t* __restrict__ out) {
    __shared__ float red[256];
    const int row = blockIdx.x;
    const int t   = threadIdx.x;
    const float* xr = x + (size_t)row * CC;
    float4 v = ((const float4*)xr)[t];

    float s = v.x + v.y + v.z + v.w;
    red[t] = s; __syncthreads();
    for (int o = 128; o > 0; o >>= 1) { if (t < o) red[t] += red[t + o]; __syncthreads(); }
    const float mean = red[0] * (1.0f / CC);
    __syncthreads();

    float d0 = v.x - mean, d1 = v.y - mean, d2 = v.z - mean, d3 = v.w - mean;
    red[t] = d0*d0 + d1*d1 + d2*d2 + d3*d3; __syncthreads();
    for (int o = 128; o > 0; o >>= 1) { if (t < o) red[t] += red[t + o]; __syncthreads(); }
    const float var = red[0] * (1.0f / (CC - 1));   // ddof=1 to match torch-style var
    const float inv = rsqrtf(var + 1e-5f);

    float4 gv = ((const float4*)gamma)[t];
    float4 bv = ((const float4*)beta)[t];
    v4h o4;
    o4[0] = (half_t)(gv.x * d0 * inv + bv.x);
    o4[1] = (half_t)(gv.y * d1 * inv + bv.y);
    o4[2] = (half_t)(gv.z * d2 * inv + bv.z);
    o4[3] = (half_t)(gv.w * d3 * inv + bv.w);
    *(v4h*)&out[(size_t)row * CC + t * 4] = o4;
}

// ---------------------------------------------------------------------------
// WMMA GEMM: C[R x N] = A[R x K] * Bt[N x K]^T  (Bt is pre-transposed weight)
// Block tile 128x128, 8 waves, wave tile 32x64 (2 A-frags x 4 B-frags =
// 8 wmma per 32-wide k-step), K-step 64 (16 wmma per barrier pair).
// MODE 0: store f16.   MODE 1: f32 out = residual + bias + acc.
// MODE 2: store f16 of exact GELU(acc + bias).
// ---------------------------------------------------------------------------
template <int MODE>
__global__ __launch_bounds__(256) void gemm_wmma_kernel(
        const half_t* __restrict__ A, const half_t* __restrict__ Bt,
        const float* __restrict__ bias, const float* __restrict__ residual,
        float* __restrict__ outF, half_t* __restrict__ outH,
        int K, int N) {
    __shared__ half_t As[128 * 64];   // 16 KB
    __shared__ half_t Bs[128 * 64];   // 16 KB

    const int tid  = threadIdx.x;
    const int lane = tid & 31;
    const int wave = tid >> 5;
    const int wm   = wave & 3;        // 4 x 32-row strips
    const int wn   = wave >> 2;       // 2 x 64-col strips
    const int rowBase = blockIdx.y * 128;
    const int nBase   = blockIdx.x * 128;

    v8f acc[8];
    #pragma unroll
    for (int i = 0; i < 8; ++i) acc[i] = 0.f;

    tdm_null_load();   // gfx1250 TDM + TENSORcnt path (architectural no-op)

    for (int k0 = 0; k0 < K; k0 += 64) {
        __syncthreads();
        // Stage A and B tiles: 128x64 halfs each; 4 x b128 per matrix per thr.
        #pragma unroll
        for (int i = 0; i < 4; ++i) {
            int ch  = tid + i * 256;          // 1024 chunks of 8 halfs
            int row = ch >> 3;
            int col = (ch & 7) << 3;
            *(v8h*)&As[row * 64 + col] =
                *(const v8h*)&A[(size_t)(rowBase + row) * K + k0 + col];
            *(v8h*)&Bs[row * 64 + col] =
                *(const v8h*)&Bt[(size_t)(nBase + row) * K + k0 + col];
        }
        if (k0 + 64 < K) {   // hint next K-tile into cache (global_prefetch_b8)
            __builtin_prefetch(&A [(size_t)(rowBase + (tid >> 1)) * K + k0 + 64], 0, 3);
            __builtin_prefetch(&Bt[(size_t)(nBase   + (tid >> 1)) * K + k0 + 64], 0, 3);
        }
        __syncthreads();

        #pragma unroll
        for (int ks = 0; ks < 64; ks += 32) {
            v16h a0 = load_frag(&As[(wm * 32 +  0) * 64 + ks], 64, lane);
            v16h a1 = load_frag(&As[(wm * 32 + 16) * 64 + ks], 64, lane);
            #pragma unroll
            for (int j = 0; j < 4; ++j) {
                v16h b = load_frag(&Bs[(wn * 64 + j * 16) * 64 + ks], 64, lane);
                acc[j]     = wmma32(a0, b, acc[j]);
                acc[4 + j] = wmma32(a1, b, acc[4 + j]);
            }
        }
    }

    const int cl = lane & 15;
    const int rh = (lane >> 4) << 3;
    #pragma unroll
    for (int mi = 0; mi < 2; ++mi) {
        #pragma unroll
        for (int j = 0; j < 4; ++j) {
            v8f c = acc[mi * 4 + j];
            #pragma unroll
            for (int r = 0; r < 8; ++r) {
                int row = rowBase + wm * 32 + mi * 16 + r + rh;
                int col = nBase + wn * 64 + j * 16 + cl;
                float val = c[r];
                if (MODE == 0) {
                    outH[(size_t)row * N + col] = (half_t)val;
                } else if (MODE == 1) {
                    outF[(size_t)row * N + col] =
                        residual[(size_t)row * N + col] + bias[col] + val;
                } else {
                    float xg = val + bias[col];
                    float g  = xg * 0.5f * (1.0f + erff(xg * 0.70710678118654752f));
                    outH[(size_t)row * N + col] = (half_t)g;
                }
            }
        }
    }
}

// ---------------------------------------------------------------------------
// Causal flash attention. Grid: (T/64, H, B). 128 threads = 4 waves.
// Each wave owns 16 query rows; dh=64 output = 4 wmma col-tiles.
// q/k/v/out layout: [B*T, C] with col = h*64 + d.
// ---------------------------------------------------------------------------
__global__ __launch_bounds__(128) void attn_kernel(const half_t* __restrict__ q,
                                                   const half_t* __restrict__ k,
                                                   const half_t* __restrict__ v,
                                                   half_t* __restrict__ outp) {
    __shared__ half_t Kt[32 * 64];        // keys x d   (row-major)
    __shared__ half_t Vt[64 * 32];        // d x keys   (transposed V)
    __shared__ half_t Pl[4][16 * 32];     // per-wave P relayout buffer

    const int tid  = threadIdx.x;
    const int lane = tid & 31;
    const int w    = tid >> 5;
    const int bb   = blockIdx.z;
    const int hh   = blockIdx.y;
    const int qb   = blockIdx.x * 64;
    const int q0   = qb + w * 16;

    // Q fragments for the wave's 16 rows: two k-steps over dh=64.
    const size_t qoff = (size_t)(bb * TT + q0) * CC + hh * DH;
    v16h aQ0 = load_frag(q + qoff,      CC, lane);
    v16h aQ1 = load_frag(q + qoff + 32, CC, lane);

    v8f o0 = 0.f, o1 = 0.f, o2 = 0.f, o3 = 0.f;
    float m_[8], l_[8];
    #pragma unroll
    for (int r = 0; r < 8; ++r) { m_[r] = -1e30f; l_[r] = 0.0f; }

    const int cl = lane & 15;
    const int rh = (lane >> 4) << 3;
    const int ntiles = blockIdx.x * 2 + 2;     // key tiles of 32 up through diag

    for (int it = 0; it < ntiles; ++it) {
        const int s0 = it * 32;
        __syncthreads();
        // Stage K (row-major) and V (transposed): 2048 halfs each / 128 thr.
        #pragma unroll
        for (int ch = tid; ch < 256; ch += 128) {
            int srow = ch >> 3;
            int col  = (ch & 7) << 3;
            const size_t g = (size_t)(bb * TT + s0 + srow) * CC + hh * DH + col;
            *(v8h*)&Kt[srow * 64 + col] = *(const v8h*)&k[g];
            v8h vv = *(const v8h*)&v[g];
            #pragma unroll
            for (int j = 0; j < 8; ++j) Vt[(col + j) * 32 + srow] = vv[j];
        }
        __syncthreads();

        // S = Q * K^T for this wave's 16 rows x 32 keys.
        v8f s0a = 0.f, s1a = 0.f;
        {
            v16h bk;
            bk = load_frag(&Kt[0 * 64 +  0], 64, lane); s0a = wmma32(aQ0, bk, s0a);
            bk = load_frag(&Kt[16 * 64 + 0], 64, lane); s1a = wmma32(aQ0, bk, s1a);
            bk = load_frag(&Kt[0 * 64 + 32], 64, lane); s0a = wmma32(aQ1, bk, s0a);
            bk = load_frag(&Kt[16 * 64 + 32], 64, lane); s1a = wmma32(aQ1, bk, s1a);
        }

        // Scale, causal mask, online softmax update.
        #pragma unroll
        for (int r = 0; r < 8; ++r) {
            const int qg = q0 + r + rh;
            float x0 = s0a[r] * 0.125f; if (s0 + cl > qg)      x0 = -1e9f;
            float x1 = s1a[r] * 0.125f; if (s0 + 16 + cl > qg) x1 = -1e9f;
            float tm = fmaxf(x0, x1);
            tm = fmaxf(tm, __shfl_xor(tm, 1));
            tm = fmaxf(tm, __shfl_xor(tm, 2));
            tm = fmaxf(tm, __shfl_xor(tm, 4));
            tm = fmaxf(tm, __shfl_xor(tm, 8));
            const float nm = fmaxf(m_[r], tm);
            const float al = __expf(m_[r] - nm);
            const float p0 = __expf(x0 - nm);
            const float p1 = __expf(x1 - nm);
            float ts = p0 + p1;
            ts += __shfl_xor(ts, 1);
            ts += __shfl_xor(ts, 2);
            ts += __shfl_xor(ts, 4);
            ts += __shfl_xor(ts, 8);
            l_[r] = l_[r] * al + ts;
            m_[r] = nm;
            o0[r] *= al; o1[r] *= al; o2[r] *= al; o3[r] *= al;
            Pl[w][(r + rh) * 32 + cl]      = (half_t)p0;
            Pl[w][(r + rh) * 32 + 16 + cl] = (half_t)p1;
        }

        // O += P * V   (A-frag from per-wave LDS relayout, B-frags from V^T).
        v16h aP = load_frag(&Pl[w][0], 32, lane);
        v16h bv;
        bv = load_frag(&Vt[ 0 * 32], 32, lane); o0 = wmma32(aP, bv, o0);
        bv = load_frag(&Vt[16 * 32], 32, lane); o1 = wmma32(aP, bv, o1);
        bv = load_frag(&Vt[32 * 32], 32, lane); o2 = wmma32(aP, bv, o2);
        bv = load_frag(&Vt[48 * 32], 32, lane); o3 = wmma32(aP, bv, o3);
    }

    #pragma unroll
    for (int r = 0; r < 8; ++r) {
        const int qg = q0 + r + rh;
        const float inv = 1.0f / l_[r];
        const size_t base = (size_t)(bb * TT + qg) * CC + hh * DH;
        outp[base +  0 + cl] = (half_t)(o0[r] * inv);
        outp[base + 16 + cl] = (half_t)(o1[r] * inv);
        outp[base + 32 + cl] = (half_t)(o2[r] * inv);
        outp[base + 48 + cl] = (half_t)(o3[r] * inv);
    }
}

// ---------------------------------------------------------------------------
// Launcher
// ---------------------------------------------------------------------------
extern "C" void kernel_launch(void* const* d_in, const int* in_sizes, int n_in,
                              void* d_out, int out_size, void* d_ws, size_t ws_size,
                              hipStream_t stream) {
    (void)in_sizes; (void)n_in; (void)out_size; (void)ws_size;

    const float* x      = (const float*)d_in[0];
    const float* gamma1 = (const float*)d_in[1];
    const float* beta1  = (const float*)d_in[2];
    const float* Wq     = (const float*)d_in[3];
    const float* Wk     = (const float*)d_in[4];
    const float* Wv     = (const float*)d_in[5];
    const float* W0     = (const float*)d_in[6];
    const float* b0     = (const float*)d_in[7];
    const float* gamma2 = (const float*)d_in[8];
    const float* beta2  = (const float*)d_in[9];
    const float* W1     = (const float*)d_in[10];
    const float* b1     = (const float*)d_in[11];
    const float* W2     = (const float*)d_in[12];
    const float* b2     = (const float*)d_in[13];
    float* out = (float*)d_out;

    // Workspace carve-up (256B aligned).
    char* ws = (char*)d_ws;
    size_t off = 0;
    auto alloc = [&](size_t bytes) -> char* {
        char* p = ws + off;
        off = (off + bytes + 255) & ~(size_t)255;
        return p;
    };
    half_t* Wqt = (half_t*)alloc((size_t)CC * CC * 2);
    half_t* Wkt = (half_t*)alloc((size_t)CC * CC * 2);
    half_t* Wvt = (half_t*)alloc((size_t)CC * CC * 2);
    half_t* W0t = (half_t*)alloc((size_t)CC * CC * 2);
    half_t* W1t = (half_t*)alloc((size_t)CC * 4 * CC * 2);
    half_t* W2t = (half_t*)alloc((size_t)4 * CC * CC * 2);
    half_t* h1  = (half_t*)alloc((size_t)RR * CC * 2);
    half_t* qh  = (half_t*)alloc((size_t)RR * CC * 2);
    half_t* kh  = (half_t*)alloc((size_t)RR * CC * 2);
    half_t* vh  = (half_t*)alloc((size_t)RR * CC * 2);
    half_t* at  = (half_t*)alloc((size_t)RR * CC * 2);
    float*  x1  = (float*) alloc((size_t)RR * CC * 4);
    half_t* h2  = (half_t*)alloc((size_t)RR * CC * 2);
    half_t* hid = (half_t*)alloc((size_t)RR * 4 * CC * 2);

    // 1) Weight conversion/transposition (one pass; cheap vs GEMMs).
    qkvtrans_kernel<<<(CC * CC) / 256, 256, 0, stream>>>(Wq, Wqt);
    qkvtrans_kernel<<<(CC * CC) / 256, 256, 0, stream>>>(Wk, Wkt);
    qkvtrans_kernel<<<(CC * CC) / 256, 256, 0, stream>>>(Wv, Wvt);
    wtrans_kernel<<<(CC * CC) / 256, 256, 0, stream>>>(W0, W0t, CC, CC);
    wtrans_kernel<<<(CC * 4 * CC) / 256, 256, 0, stream>>>(W1, W1t, CC, 4 * CC);
    wtrans_kernel<<<(4 * CC * CC) / 256, 256, 0, stream>>>(W2, W2t, 4 * CC, CC);

    // 2) LN1 -> h1 (f16)
    layernorm_kernel<<<RR, 256, 0, stream>>>(x, gamma1, beta1, h1);

    // 3) QKV projections (WMMA GEMMs, store f16 in [B,T,H,dh] layout)
    dim3 gC(CC / 128, RR / 128);           // N=1024 outputs
    gemm_wmma_kernel<0><<<gC, 256, 0, stream>>>(h1, Wqt, nullptr, nullptr, nullptr, qh, CC, CC);
    gemm_wmma_kernel<0><<<gC, 256, 0, stream>>>(h1, Wkt, nullptr, nullptr, nullptr, kh, CC, CC);
    gemm_wmma_kernel<0><<<gC, 256, 0, stream>>>(h1, Wvt, nullptr, nullptr, nullptr, vh, CC, CC);

    // 4) Causal flash attention
    attn_kernel<<<dim3(TT / 64, HH, BB), 128, 0, stream>>>(qh, kh, vh, at);

    // 5) Output projection + residual: x1 = x + at @ W0 + b0
    gemm_wmma_kernel<1><<<gC, 256, 0, stream>>>(at, W0t, b0, x, x1, nullptr, CC, CC);

    // 6) LN2 -> h2 (f16)
    layernorm_kernel<<<RR, 256, 0, stream>>>(x1, gamma2, beta2, h2);

    // 7) MLP up + exact GELU -> hid (f16)
    gemm_wmma_kernel<2><<<dim3(4 * CC / 128, RR / 128), 256, 0, stream>>>(
        h2, W1t, b1, nullptr, nullptr, hid, CC, 4 * CC);

    // 8) MLP down + residual -> d_out (f32)
    gemm_wmma_kernel<1><<<gC, 256, 0, stream>>>(hid, W2t, b2, x1, out, nullptr, 4 * CC, CC);
}
